// DeformableAttention_19129784336523
// MI455X (gfx1250) — compile-verified, matrix operand order checked by reading, not run on hardware
//
#include <hip/hip_runtime.h>
#include <hip/hip_bf16.h>

typedef __attribute__((ext_vector_type(16))) __bf16 v16bf;
typedef __attribute__((ext_vector_type(8)))  __bf16 v8bf;
typedef __attribute__((ext_vector_type(8)))  float  v8f;

#define B_  16
#define Q_  900
#define D_  256
#define NH_ 8
#define P_  4
#define H_  100
#define W_  100
#define HW_ 10000
#define DH_ 32

// ---------------------------------------------------------------------------
// WMMA 16x16x32 bf16 fragment helpers.
// Per ISA 7.12.2 (16-bit A 16x32): lane half h holds K in [8h,8h+8) U [16+8h,16+8h+8),
// i.e. two contiguous 8-element runs -> wide loads.
// ---------------------------------------------------------------------------

// A fragment from an fp32 row-major row (convert to bf16): 4x b128 loads.
__device__ __forceinline__ v16bf load_a_f32(const float* __restrict__ rowbase, int half) {
  float4 a0 = *(const float4*)(rowbase + 8 * half);
  float4 a1 = *(const float4*)(rowbase + 8 * half + 4);
  float4 a2 = *(const float4*)(rowbase + 16 + 8 * half);
  float4 a3 = *(const float4*)(rowbase + 16 + 8 * half + 4);
  v16bf a;
  a[0]  = (__bf16)a0.x; a[1]  = (__bf16)a0.y; a[2]  = (__bf16)a0.z; a[3]  = (__bf16)a0.w;
  a[4]  = (__bf16)a1.x; a[5]  = (__bf16)a1.y; a[6]  = (__bf16)a1.z; a[7]  = (__bf16)a1.w;
  a[8]  = (__bf16)a2.x; a[9]  = (__bf16)a2.y; a[10] = (__bf16)a2.z; a[11] = (__bf16)a2.w;
  a[12] = (__bf16)a3.x; a[13] = (__bf16)a3.y; a[14] = (__bf16)a3.z; a[15] = (__bf16)a3.w;
  return a;
}

// A fragment from a bf16 row-major row: 2x b128 loads + register concat.
__device__ __forceinline__ v16bf load_a_bf16(const unsigned short* __restrict__ rowbase, int half) {
  v8bf lo = *(const v8bf*)(rowbase + 8 * half);
  v8bf hi = *(const v8bf*)(rowbase + 16 + 8 * half);
  return __builtin_shufflevector(lo, hi, 0, 1, 2, 3, 4, 5, 6, 7,
                                         8, 9, 10, 11, 12, 13, 14, 15);
}

// B fragment from fragment-major LDS: each lane's 16 halfs are contiguous 32B
// -> exactly 2x ds_load_b128, zero per-fragment address math beyond one add.
__device__ __forceinline__ v16bf load_b_frag(const unsigned short* sF, int frag, int lane) {
  return *(const v16bf*)(sF + (frag * 32 + lane) * 16);
}

// Stage a 64-wide K chunk of a [256,256] fp32 weight matrix into LDS in
// fragment-major bf16 layout (32 fragments x 32 lanes x 16 halfs = 32KB).
// Two k-consecutive elements pack into one u32 store.
__device__ __forceinline__ void stage_w_chunk(unsigned short* sF,
                                              const float* __restrict__ Wsrc,
                                              int k0, int tid) {
  for (int i = tid; i < 32 * 256; i += 256) {
    int m  = i >> 8;          // kk pair index: kk = 2m, 2m+1
    int n  = i & 255;
    int kk = 2 * m;
    float f0 = Wsrc[(k0 + kk) * 256 + n];
    float f1 = Wsrc[(k0 + kk + 1) * 256 + n];
    int kis  = kk & 31;                       // k within 32-wide WMMA step
    int ks   = kk >> 5;                       // which of 2 steps in chunk
    int half = (kis >> 3) & 1;                // lane half owning this k
    int hi   = (kis >> 4) & 1;                // upper 16 of the step
    int j    = ((kis - 16 * hi - 8 * half) >> 1) + 4 * hi;  // VGPR pair slot
    int lane = (n & 15) + 16 * half;
    int frag = ks * 16 + (n >> 4);
    __bf16 b0 = (__bf16)f0, b1 = (__bf16)f1;
    unsigned int pk = (unsigned int)__builtin_bit_cast(unsigned short, b0)
                    | ((unsigned int)__builtin_bit_cast(unsigned short, b1) << 16);
    *(unsigned int*)(sF + (frag * 32 + lane) * 16 + 2 * j) = pk;
  }
}

// ---------------------------------------------------------------------------
// K1: per-query sampling offsets + softmaxed attention weights.
// ---------------------------------------------------------------------------
__global__ __launch_bounds__(128) void k_sampling(
    const float* __restrict__ q, const float* __restrict__ ref,
    const float* __restrict__ Woff, const float* __restrict__ boff,
    const float* __restrict__ Wat,  const float* __restrict__ bat,
    float* __restrict__ locs, float* __restrict__ attnw)
{
  __shared__ float qv[256];
  __shared__ float slog[32];
  const int bq = blockIdx.x;
  const int t  = threadIdx.x;
  qv[t]       = q[bq * 256 + t];
  qv[t + 128] = q[bq * 256 + 128 + t];
  __syncthreads();

  if (t < 64) {                       // sampling offsets (NH*P*2 = 64 outputs)
    float s = boff[t];
    for (int k = 0; k < 256; ++k) s += qv[k] * Woff[k * 64 + t];
    int h = t >> 3, p = (t >> 1) & 3, c = t & 1;
    float r   = ref[bq * 2 + c];
    float loc = fminf(fmaxf(r + s, 0.f), 1.f);
    float sc  = loc * (float)((c == 0) ? (W_ - 1) : (H_ - 1));
    int base  = (bq * NH_ + h) * P_ + p;
    locs[base * 2 + c] = sc;
  } else if (t < 96) {                // attention logits (NH*P = 32 outputs)
    int j = t - 64;
    float s = bat[j];
    for (int k = 0; k < 256; ++k) s += qv[k] * Wat[k * 32 + j];
    slog[j] = s;
  }
  __syncthreads();

  if (t < 32) {                       // softmax over P=4 per head
    int h = t >> 2;
    float l0 = slog[h * 4 + 0], l1 = slog[h * 4 + 1];
    float l2 = slog[h * 4 + 2], l3 = slog[h * 4 + 3];
    float m  = fmaxf(fmaxf(l0, l1), fmaxf(l2, l3));
    float e0 = __expf(l0 - m), e1 = __expf(l1 - m);
    float e2 = __expf(l2 - m), e3 = __expf(l3 - m);
    float inv = 1.f / (e0 + e1 + e2 + e3);
    int   pp  = t & 3;
    float mine = (pp == 0) ? e0 : (pp == 1) ? e1 : (pp == 2) ? e2 : e3;
    attnw[bq * 32 + t] = mine * inv;
  }
}

// ---------------------------------------------------------------------------
// K2: value projection GEMM [160000,256] x [256,256] -> bf16 values laid out
// [B][NH][HW][DH] for coalesced per-head gather. bf16 WMMA, f32 accumulate.
// 8 waves/block; wave owns 16 rows x all 256 cols (A loaded once per row).
// ---------------------------------------------------------------------------
__global__ __launch_bounds__(256) void k_value_proj(
    const float* __restrict__ X, const float* __restrict__ Wv,
    const float* __restrict__ bv, unsigned short* __restrict__ vals)
{
  __shared__ unsigned short sF[64 * 256];     // 32KB fragment-major chunk
  const int tid  = threadIdx.x;
  const int wave = tid >> 5;
  const int lane = tid & 31;
  const int half = lane >> 4;
  const int mrow = lane & 15;
  const int mt   = blockIdx.x * 8 + wave;     // 0..9999 (exact)
  const int rowA = mt * 16 + mrow;

  v8f zero = {0.f, 0.f, 0.f, 0.f, 0.f, 0.f, 0.f, 0.f};
  v8f acc[16];
#pragma unroll
  for (int t = 0; t < 16; ++t) acc[t] = zero;

  for (int cc = 0; cc < 4; ++cc) {            // K chunks of 64
    __syncthreads();
    stage_w_chunk(sF, Wv, cc * 64, tid);
    __syncthreads();
#pragma unroll
    for (int ks = 0; ks < 2; ++ks) {          // 2 x K=32 WMMA steps per chunk
      v16bf afrag = load_a_f32(X + rowA * 256 + cc * 64 + ks * 32, half);
#pragma unroll
      for (int nt = 0; nt < 16; ++nt) {
        v16bf bfrag = load_b_frag(sF, ks * 16 + nt, lane);
        acc[nt] = __builtin_amdgcn_wmma_f32_16x16x32_bf16(
            false, afrag, false, bfrag, (short)0, acc[nt], false, false);
      }
    }
  }
  // store with bias, scattered to [B][NH][HW][DH] bf16
#pragma unroll
  for (int r = 0; r < 8; ++r) {
    int gRow = mt * 16 + r + 8 * half;
    int b  = gRow / HW_;
    int hw = gRow % HW_;
#pragma unroll
    for (int nt = 0; nt < 16; ++nt) {
      int col = nt * 16 + mrow;
      float o = acc[nt][r] + bv[col];
      int h = col >> 5, dc = col & 31;
      __bf16 hb = (__bf16)o;
      vals[((b * NH_ + h) * HW_ + hw) * DH_ + dc] = __builtin_bit_cast(unsigned short, hb);
    }
  }
}

// ---------------------------------------------------------------------------
// K3: bilinear gather + attention-weighted sum. One block per (b,q);
// thread = (head, channel). Corner loads are coalesced 64B bf16 lines,
// served from L2 (82MB value tensor < 192MB L2). Output bf16 [14400,256].
// ---------------------------------------------------------------------------
__global__ __launch_bounds__(256) void k_gather(
    const unsigned short* __restrict__ vals, const float* __restrict__ locs,
    const float* __restrict__ attnw, unsigned short* __restrict__ interp)
{
  const int bq = blockIdx.x;
  const int b  = bq / Q_;
  const int t  = threadIdx.x;
  const int h  = t >> 5, dc = t & 31;
  const int vbase = (b * NH_ + h) * HW_;
  float acc = 0.f;
#pragma unroll
  for (int p = 0; p < P_; ++p) {
    int sb = (bq * NH_ + h) * P_ + p;
    float aw = attnw[sb];
    float sx = locs[sb * 2 + 0];
    float sy = locs[sb * 2 + 1];
    int x0 = min(max((int)floorf(sx), 0), W_ - 1);
    int y0 = min(max((int)floorf(sy), 0), H_ - 1);
    int x1 = min(x0 + 1, W_ - 1);
    int y1 = min(y0 + 1, H_ - 1);
    float wx1 = sx - (float)x0, wx0 = 1.f - wx1;
    float wy1 = sy - (float)y0, wy0 = 1.f - wy1;
    float v00 = (float)__builtin_bit_cast(__bf16, vals[(vbase + y0 * W_ + x0) * DH_ + dc]);
    float v01 = (float)__builtin_bit_cast(__bf16, vals[(vbase + y0 * W_ + x1) * DH_ + dc]);
    float v10 = (float)__builtin_bit_cast(__bf16, vals[(vbase + y1 * W_ + x0) * DH_ + dc]);
    float v11 = (float)__builtin_bit_cast(__bf16, vals[(vbase + y1 * W_ + x1) * DH_ + dc]);
    float s = wx0 * wy0 * v00 + wx1 * wy0 * v01 + wx0 * wy1 * v10 + wx1 * wy1 * v11;
    acc += aw * s;
  }
  __bf16 hb = (__bf16)acc;
  interp[bq * 256 + t] = __builtin_bit_cast(unsigned short, hb);
}

// ---------------------------------------------------------------------------
// K4: output projection [14400,256]bf16 x [256,256] + bias -> f32 out.
// Same WMMA structure as K2.
// ---------------------------------------------------------------------------
__global__ __launch_bounds__(256) void k_out_proj(
    const unsigned short* __restrict__ Xi, const float* __restrict__ Wo,
    const float* __restrict__ bo, float* __restrict__ out)
{
  __shared__ unsigned short sF[64 * 256];
  const int tid  = threadIdx.x;
  const int wave = tid >> 5;
  const int lane = tid & 31;
  const int half = lane >> 4;
  const int mrow = lane & 15;
  const int mt   = blockIdx.x * 8 + wave;     // 0..903, valid < 900
  const bool valid = (mt < 900);
  const int rowA = mt * 16 + mrow;

  v8f zero = {0.f, 0.f, 0.f, 0.f, 0.f, 0.f, 0.f, 0.f};
  v8f acc[16];
#pragma unroll
  for (int t = 0; t < 16; ++t) acc[t] = zero;

  for (int cc = 0; cc < 4; ++cc) {
    __syncthreads();
    stage_w_chunk(sF, Wo, cc * 64, tid);
    __syncthreads();
    if (valid) {
#pragma unroll
      for (int ks = 0; ks < 2; ++ks) {
        v16bf afrag = load_a_bf16(Xi + rowA * 256 + cc * 64 + ks * 32, half);
#pragma unroll
        for (int nt = 0; nt < 16; ++nt) {
          v16bf bfrag = load_b_frag(sF, ks * 16 + nt, lane);
          acc[nt] = __builtin_amdgcn_wmma_f32_16x16x32_bf16(
              false, afrag, false, bfrag, (short)0, acc[nt], false, false);
        }
      }
    }
  }
  if (valid) {
#pragma unroll
    for (int r = 0; r < 8; ++r) {
      int gRow = mt * 16 + r + 8 * half;
#pragma unroll
      for (int nt = 0; nt < 16; ++nt) {
        int col = nt * 16 + mrow;
        out[gRow * 256 + col] = acc[nt][r] + bo[col];
      }
    }
  }
}

extern "C" void kernel_launch(void* const* d_in, const int* in_sizes, int n_in,
                              void* d_out, int out_size, void* d_ws, size_t ws_size,
                              hipStream_t stream) {
  const float* q    = (const float*)d_in[0];
  const float* ref  = (const float*)d_in[1];
  const float* xin  = (const float*)d_in[2];
  // d_in[3] = h, d_in[4] = w (fixed 100x100, hardcoded)
  const float* Woff = (const float*)d_in[5];
  const float* boff = (const float*)d_in[6];
  const float* Wat  = (const float*)d_in[7];
  const float* bat  = (const float*)d_in[8];
  const float* Wv   = (const float*)d_in[9];
  const float* bv   = (const float*)d_in[10];
  const float* Wo   = (const float*)d_in[11];
  const float* bo   = (const float*)d_in[12];
  float* out = (float*)d_out;
  (void)in_sizes; (void)n_in; (void)out_size; (void)ws_size;

  // workspace carve-out
  char* ws = (char*)d_ws;
  size_t off = 0;
  auto carve = [&](size_t bytes) {
    void* p = ws + off;
    off += bytes;
    off = (off + 255) & ~(size_t)255;
    return p;
  };
  unsigned short* vals   = (unsigned short*)carve((size_t)B_ * NH_ * HW_ * DH_ * 2); // 81.9 MB
  unsigned short* interp = (unsigned short*)carve((size_t)B_ * Q_ * D_ * 2);         // 7.4 MB
  float* locs  = (float*)carve((size_t)B_ * Q_ * NH_ * P_ * 2 * 4);                  // 3.7 MB
  float* attnw = (float*)carve((size_t)B_ * Q_ * NH_ * P_ * 4);                      // 1.8 MB

  k_sampling  <<<B_ * Q_, 128, 0, stream>>>(q, ref, Woff, boff, Wat, bat, locs, attnw);
  k_value_proj<<<1250,    256, 0, stream>>>(xin, Wv, bv, vals);
  k_gather    <<<B_ * Q_, 256, 0, stream>>>(vals, locs, attnw, interp);
  k_out_proj  <<<113,     256, 0, stream>>>(interp, Wo, bo, out);
}
